// Mamba_21844203668182
// MI455X (gfx1250) — matrix operand order repaired and checked by reading, not compiled, not run on hardware
//
#include <hip/hip_runtime.h>
#include <hip/hip_bf16.h>
#include <math.h>

// ---------------- model dims ----------------
#define BATCH   2
#define SEQ     1024
#define DMODEL  768
#define NLAYER  4
#define DINNER  1536
#define DSTATE  16
#define DTRANK  48
#define DCONV   4
#define VOCAB   50280
#define ROWS    (BATCH*SEQ)          // 2048
#define XPROJ_N (DTRANK + 2*DSTATE)  // 80

typedef __attribute__((ext_vector_type(16))) __bf16 bf16x16;
typedef __attribute__((ext_vector_type(8)))  __bf16 bf16x8;
typedef __attribute__((ext_vector_type(8)))  float  f32x8;
typedef __attribute__((ext_vector_type(4)))  float  f32x4;

// GEMM block tile
#define BM 64
#define BN 64
#define BK 32
#define LDSK 48   // padded row stride in bf16 (96 B, keeps ds_*_b128 16B-aligned)

// ---------------- embedding gather ----------------
__global__ void embed_k(const int* __restrict__ ids, const float* __restrict__ emb,
                        float* __restrict__ x)
{
    int i = blockIdx.x * 256 + threadIdx.x;            // over ROWS*DMODEL
    if (i >= ROWS * DMODEL) return;
    int t = i / DMODEL, d = i % DMODEL;
    x[i] = emb[(size_t)ids[t] * DMODEL + d];
}

// ---------------- rmsnorm (one block per row) ----------------
__global__ __launch_bounds__(256)
void rmsnorm_k(const float* __restrict__ x, const float* __restrict__ w,
               float* __restrict__ o, int d)
{
    int row = blockIdx.x;
    const float* xr = x + (size_t)row * d;
    float s = 0.f;
    for (int i = threadIdx.x; i < d; i += 256) s += xr[i] * xr[i];
    __shared__ float red[256];
    red[threadIdx.x] = s;
    __syncthreads();
    for (int off = 128; off > 0; off >>= 1) {
        if (threadIdx.x < off) red[threadIdx.x] += red[threadIdx.x + off];
        __syncthreads();
    }
    float inv = rsqrtf(red[0] / (float)d + 1e-5f);
    for (int i = threadIdx.x; i < d; i += 256)
        o[(size_t)row * d + i] = xr[i] * inv * w[i];
}

// ---------------- bf16 WMMA GEMM: C[M,N] = A[M,K] @ B  (BT: B is [N,K]) ----------------
// 128 threads = 4 waves; block tile 64x64; each wave computes a 32x32 quadrant
// as 2x2 WMMA tiles (4 x v_wmma_f32_16x16x32_bf16 per K-step, fragment reuse).
template<bool BT>
__global__ __launch_bounds__(128)
void gemm_wmma(const float* __restrict__ A, int lda,
               const float* __restrict__ B, int ldb,
               float* __restrict__ C, int ldc,
               int M, int N, int K)
{
    __shared__ __bf16 As[BM][LDSK];   // [m][k]
    __shared__ __bf16 Bs[BN][LDSK];   // [n][k] (B held transposed in LDS)

    const int tid  = threadIdx.x;
    const int wave = tid >> 5;
    const int lane = tid & 31;
    const int wr = wave >> 1, wc = wave & 1;   // 2x2 wave grid
    const int mb = blockIdx.y * BM;
    const int nb = blockIdx.x * BN;
    const int hf = lane >> 4;                  // half-wave 0/1
    const int lr = lane & 15;

    f32x8 acc00 = {}, acc01 = {}, acc10 = {}, acc11 = {};

    for (int k0 = 0; k0 < K; k0 += BK) {
        const bool full = (k0 + BK <= K) && (nb + BN <= N) && (mb + BM <= M);

        if (full) {
            // ---- A tile: 16 contiguous floats/thread -> 2 x ds_store_b128 ----
            {
                int r  = tid >> 1;
                int c0 = (tid & 1) * 16;
                const float* ap = A + (size_t)(mb + r) * lda + k0 + c0;
                bf16x8 v0, v1;
                #pragma unroll
                for (int i = 0; i < 8; ++i) { v0[i] = (__bf16)ap[i]; v1[i] = (__bf16)ap[8 + i]; }
                *reinterpret_cast<bf16x8*>(&As[r][c0])     = v0;
                *reinterpret_cast<bf16x8*>(&As[r][c0 + 8]) = v1;
            }
            // ---- B tile (transposed into Bs[n][k]) ----
            if (BT) {
                int n  = tid >> 1;
                int c0 = (tid & 1) * 16;
                const float* bp = B + (size_t)(nb + n) * ldb + k0 + c0;
                bf16x8 v0, v1;
                #pragma unroll
                for (int i = 0; i < 8; ++i) { v0[i] = (__bf16)bp[i]; v1[i] = (__bf16)bp[8 + i]; }
                *reinterpret_cast<bf16x8*>(&Bs[n][c0])     = v0;
                *reinterpret_cast<bf16x8*>(&Bs[n][c0 + 8]) = v1;
            } else {
                // lane-coalesced column loads, contiguous LDS writes
                int n  = tid & 63;
                int kh = (tid >> 6) * 16;
                const float* bp = B + (size_t)(k0 + kh) * ldb + nb + n;
                bf16x8 v0, v1;
                #pragma unroll
                for (int i = 0; i < 8; ++i) { v0[i] = (__bf16)bp[(size_t)i * ldb];
                                              v1[i] = (__bf16)bp[(size_t)(i + 8) * ldb]; }
                *reinterpret_cast<bf16x8*>(&Bs[n][kh])     = v0;
                *reinterpret_cast<bf16x8*>(&Bs[n][kh + 8]) = v1;
            }
        } else {
            // ---- rare edge tiles: guarded element loads ----
            {
                int r  = tid >> 1;
                int c0 = (tid & 1) * 16;
                int gm = mb + r;
                #pragma unroll
                for (int i = 0; i < 16; ++i) {
                    int gk = k0 + c0 + i;
                    float v = (gm < M && gk < K) ? A[(size_t)gm * lda + gk] : 0.f;
                    As[r][c0 + i] = (__bf16)v;
                }
            }
            if (BT) {
                int n  = tid >> 1;
                int c0 = (tid & 1) * 16;
                int gn = nb + n;
                #pragma unroll
                for (int i = 0; i < 16; ++i) {
                    int gk = k0 + c0 + i;
                    float v = (gn < N && gk < K) ? B[(size_t)gn * ldb + gk] : 0.f;
                    Bs[n][c0 + i] = (__bf16)v;
                }
            } else {
                int n  = tid & 63;
                int kh = (tid >> 6) * 16;
                int gn = nb + n;
                #pragma unroll
                for (int i = 0; i < 16; ++i) {
                    int gk = k0 + kh + i;
                    float v = (gn < N && gk < K) ? B[(size_t)gk * ldb + gn] : 0.f;
                    Bs[n][kh + i] = (__bf16)v;
                }
            }
        }
        __syncthreads();

        // ---- build CDNA5 wave32 fragments (2 A rows, 2 B cols) ----
        bf16x16 a0, a1, b0, b1;
        const int ar0 = wr * 32 + lr, ar1 = ar0 + 16;
        const int bc0 = wc * 32 + lr, bc1 = bc0 + 16;
        #pragma unroll
        for (int e = 0; e < 16; ++e) {
            int ka = 16 * hf + e;                       // A: 16x32 bf16, k = 16*half + e (matches B)
            a0[e] = As[ar0][ka];
            a1[e] = As[ar1][ka];
            b0[e] = Bs[bc0][16 * hf + e];               // B: 32x16, N = lane&15
            b1[e] = Bs[bc1][16 * hf + e];
        }
        acc00 = __builtin_amdgcn_wmma_f32_16x16x32_bf16(false, a0, false, b0, (short)0, acc00, false, false);
        acc01 = __builtin_amdgcn_wmma_f32_16x16x32_bf16(false, a0, false, b1, (short)0, acc01, false, false);
        acc10 = __builtin_amdgcn_wmma_f32_16x16x32_bf16(false, a1, false, b0, (short)0, acc10, false, false);
        acc11 = __builtin_amdgcn_wmma_f32_16x16x32_bf16(false, a1, false, b1, (short)0, acc11, false, false);
        __syncthreads();
    }

    // ---- store: lane holds N = lane&15 (+16), VGPR r -> M = r + 8*half ----
    const int n0 = nb + wc * 32 + lr, n1 = n0 + 16;
    #pragma unroll
    for (int r = 0; r < 8; ++r) {
        int m0 = mb + wr * 32 + r + 8 * hf, m1 = m0 + 16;
        if (m0 < M && n0 < N) C[(size_t)m0 * ldc + n0] = acc00[r];
        if (m0 < M && n1 < N) C[(size_t)m0 * ldc + n1] = acc01[r];
        if (m1 < M && n0 < N) C[(size_t)m1 * ldc + n0] = acc10[r];
        if (m1 < M && n1 < N) C[(size_t)m1 * ldc + n1] = acc11[r];
    }
}

// ---------------- depthwise causal conv (k=4) + SiLU; input = first half of xr ----------------
__global__ void conv_silu_k(const float* __restrict__ xr, const float* __restrict__ cw,
                            const float* __restrict__ cb, float* __restrict__ xs)
{
    int i = blockIdx.x * 256 + threadIdx.x;            // over BATCH*SEQ*DINNER
    if (i >= BATCH * SEQ * DINNER) return;
    int c = i % DINNER;
    int t = (i / DINNER) % SEQ;
    int b = i / (DINNER * SEQ);
    float v = cb[c];
    #pragma unroll
    for (int j = 0; j < DCONV; ++j) {
        int t2 = t - (DCONV - 1) + j;
        if (t2 >= 0)
            v += cw[c * DCONV + j] * xr[((size_t)(b * SEQ + t2)) * (2 * DINNER) + c];
    }
    xs[i] = v / (1.f + __expf(-v));                    // silu
}

// ---------------- dt = softplus(dt_raw + bias) ----------------
__global__ void softplus_k(float* __restrict__ dlt, const float* __restrict__ dtb)
{
    int i = blockIdx.x * 256 + threadIdx.x;            // over ROWS*DINNER
    if (i >= ROWS * DINNER) return;
    float z = dlt[i] + dtb[i % DINNER];
    dlt[i] = (z > 20.f) ? z : log1pf(__expf(z));
}

// ---------------- selective scan: one thread per (b,d), 16-state in regs ----------------
__global__ __launch_bounds__(256)
void scan_k(const float* __restrict__ u, const float* __restrict__ dlt,
            const float* __restrict__ xdbl,     // rows of 80: [48 dt | 16 B | 16 C]
            const float* __restrict__ A_log, const float* __restrict__ Dv,
            float* __restrict__ y)
{
    const int blocksPerB = DINNER / 256;               // 6
    int b = blockIdx.x / blocksPerB;
    int d = (blockIdx.x % blocksPerB) * 256 + threadIdx.x;

    float An[DSTATE], st[DSTATE];
    #pragma unroll
    for (int n = 0; n < DSTATE; ++n) {
        An[n] = -__expf(A_log[(size_t)d * DSTATE + n]);
        st[n] = 0.f;
    }
    float Dd = Dv[d];

    __shared__ float sB[DSTATE], sC[DSTATE];
    for (int t = 0; t < SEQ; ++t) {
        size_t row80 = ((size_t)b * SEQ + t) * XPROJ_N;
        if (threadIdx.x < DSTATE)
            sB[threadIdx.x] = xdbl[row80 + DTRANK + threadIdx.x];
        else if (threadIdx.x < 2 * DSTATE)
            sC[threadIdx.x - DSTATE] = xdbl[row80 + DTRANK + DSTATE + (threadIdx.x - DSTATE)];
        __syncthreads();

        size_t idx = ((size_t)b * SEQ + t) * DINNER + d;
        float dl = dlt[idx];
        float uv = u[idx];
        float yv = 0.f;
        #pragma unroll
        for (int n = 0; n < DSTATE; ++n) {
            float dA = __expf(dl * An[n]);
            st[n] = dA * st[n] + dl * sB[n] * uv;
            yv += st[n] * sC[n];
        }
        y[idx] = yv + uv * Dd;
        __syncthreads();
    }
}

// ---------------- gated = y * silu(res), res = second half of xr ----------------
__global__ void gate_silu_k(const float* __restrict__ y, const float* __restrict__ xr,
                            float* __restrict__ yg)
{
    int i = blockIdx.x * 256 + threadIdx.x;            // over ROWS*DINNER
    if (i >= ROWS * DINNER) return;
    int t = i / DINNER, d = i % DINNER;
    float r = xr[(size_t)t * (2 * DINNER) + DINNER + d];
    yg[i] = y[i] * (r / (1.f + __expf(-r)));
}

// ---------------- residual add ----------------
__global__ void addres_k(float* __restrict__ x, const float* __restrict__ t)
{
    int i = blockIdx.x * 256 + threadIdx.x;            // over ROWS*DMODEL
    if (i >= ROWS * DMODEL) return;
    x[i] += t[i];
}

// ---------------- host orchestration ----------------
static inline int cdiv(int a, int b) { return (a + b - 1) / b; }

extern "C" void kernel_launch(void* const* d_in, const int* in_sizes, int n_in,
                              void* d_out, int out_size, void* d_ws, size_t ws_size,
                              hipStream_t stream)
{
    (void)in_sizes; (void)n_in; (void)out_size; (void)ws_size;

    const int*   ids    = (const int*)  d_in[0];
    const float* emb    = (const float*)d_in[1];
    const float* in_w   = (const float*)d_in[2];   // [L,768,3072]
    const float* conv_w = (const float*)d_in[3];   // [L,1536,4]
    const float* conv_b = (const float*)d_in[4];   // [L,1536]
    const float* xpw    = (const float*)d_in[5];   // [L,1536,80]
    const float* dtw    = (const float*)d_in[6];   // [L,48,1536]
    const float* dtb    = (const float*)d_in[7];   // [L,1536]
    const float* A_log  = (const float*)d_in[8];   // [L,1536,16]
    const float* Dp     = (const float*)d_in[9];   // [L,1536]
    const float* out_w  = (const float*)d_in[10];  // [L,1536,768]
    const float* norm_w = (const float*)d_in[11];  // [L,768]
    const float* normf  = (const float*)d_in[12];  // [768]
    float* logits = (float*)d_out;

    // workspace layout (fp32)
    float* ws   = (float*)d_ws;
    float* x    = ws;                               // ROWS*DMODEL
    float* xn   = x    + (size_t)ROWS * DMODEL;     // ROWS*DMODEL
    float* xr   = xn   + (size_t)ROWS * DMODEL;     // ROWS*2*DINNER
    float* xs   = xr   + (size_t)ROWS * 2 * DINNER; // ROWS*DINNER
    float* xdbl = xs   + (size_t)ROWS * DINNER;     // ROWS*80
    float* dlt  = xdbl + (size_t)ROWS * XPROJ_N;    // ROWS*DINNER
    float* yb   = dlt  + (size_t)ROWS * DINNER;     // ROWS*DINNER

    const dim3 blk256(256), blk128(128);
    const int nEmb = ROWS * DMODEL, nInn = ROWS * DINNER;

    // 1. embedding gather
    embed_k<<<cdiv(nEmb, 256), blk256, 0, stream>>>(ids, emb, x);

    // 2. layers
    for (int L = 0; L < NLAYER; ++L) {
        const float* in_wL  = in_w   + (size_t)L * DMODEL * 2 * DINNER;
        const float* cwL    = conv_w + (size_t)L * DINNER * DCONV;
        const float* cbL    = conv_b + (size_t)L * DINNER;
        const float* xpwL   = xpw    + (size_t)L * DINNER * XPROJ_N;
        const float* dtwL   = dtw    + (size_t)L * DTRANK * DINNER;
        const float* dtbL   = dtb    + (size_t)L * DINNER;
        const float* AlogL  = A_log  + (size_t)L * DINNER * DSTATE;
        const float* DL     = Dp     + (size_t)L * DINNER;
        const float* out_wL = out_w  + (size_t)L * DINNER * DMODEL;
        const float* nwL    = norm_w + (size_t)L * DMODEL;

        // rmsnorm -> xn
        rmsnorm_k<<<ROWS, blk256, 0, stream>>>(x, nwL, xn, DMODEL);

        // in_proj: xr = xn @ in_wL   (2048x768 @ 768x3072)
        gemm_wmma<false><<<dim3(cdiv(2 * DINNER, BN), ROWS / BM), blk128, 0, stream>>>(
            xn, DMODEL, in_wL, 2 * DINNER, xr, 2 * DINNER, ROWS, 2 * DINNER, DMODEL);

        // conv + silu -> xs
        conv_silu_k<<<cdiv(nInn, 256), blk256, 0, stream>>>(xr, cwL, cbL, xs);

        // x_proj: xdbl = xs @ xpwL   (2048x1536 @ 1536x80)
        gemm_wmma<false><<<dim3(cdiv(XPROJ_N, BN), ROWS / BM), blk128, 0, stream>>>(
            xs, DINNER, xpwL, XPROJ_N, xdbl, XPROJ_N, ROWS, XPROJ_N, DINNER);

        // dt_proj: dlt = xdbl[:, :48] @ dtwL   (A row-stride 80, K=48)
        gemm_wmma<false><<<dim3(cdiv(DINNER, BN), ROWS / BM), blk128, 0, stream>>>(
            xdbl, XPROJ_N, dtwL, DINNER, dlt, DINNER, ROWS, DINNER, DTRANK);

        // dlt = softplus(dlt + bias)
        softplus_k<<<cdiv(nInn, 256), blk256, 0, stream>>>(dlt, dtbL);

        // selective scan -> yb (includes + u*D)
        scan_k<<<BATCH * (DINNER / 256), blk256, 0, stream>>>(xs, dlt, xdbl, AlogL, DL, yb);

        // gated = yb * silu(res)  (reuse xs)
        gate_silu_k<<<cdiv(nInn, 256), blk256, 0, stream>>>(yb, xr, xs);

        // out_proj: xn = gated @ out_wL   (2048x1536 @ 1536x768)
        gemm_wmma<false><<<dim3(cdiv(DMODEL, BN), ROWS / BM), blk128, 0, stream>>>(
            xs, DINNER, out_wL, DMODEL, xn, DMODEL, ROWS, DMODEL, DINNER);

        // residual
        addres_k<<<cdiv(nEmb, 256), blk256, 0, stream>>>(x, xn);
    }

    // 3. final rmsnorm
    rmsnorm_k<<<ROWS, blk256, 0, stream>>>(x, normf, xn, DMODEL);

    // 4. logits = xn @ emb^T   (2048x768 @ (50280x768)^T) — emb stays L2-resident
    gemm_wmma<true><<<dim3(cdiv(VOCAB, BN), ROWS / BM), blk128, 0, stream>>>(
        xn, DMODEL, emb, DMODEL, logits, VOCAB, ROWS, VOCAB, DMODEL);
}
